// GroupedQueryAttention_47717086658660
// MI455X (gfx1250) — matrix operand order repaired and checked by reading, not compiled, not run on hardware
//
#include <hip/hip_runtime.h>
#include <hip/hip_bf16.h>
#include <math.h>

typedef __bf16 bf16_t;
typedef __attribute__((ext_vector_type(16))) __bf16 v16bf;
typedef __attribute__((ext_vector_type(8)))  __bf16 v8bf;
typedef __attribute__((ext_vector_type(8)))  float  v8f;

#define NQH   16
#define NKVH  4
#define EMB_  2048
#define HEAD_ 128
#define WIN_  1024
#define SEQ_  2048
#define BATCH_ 2
#define MTOT_ (BATCH_*SEQ_)   /* 4096 */

static __device__ __forceinline__ bf16_t f2bf(float f) {
  union { float f; unsigned u; } a; a.f = f;
  unsigned r = a.u + 0x7FFFu + ((a.u >> 16) & 1u);
  union { unsigned short s; bf16_t b; } c; c.s = (unsigned short)(r >> 16);
  return c.b;
}
static __device__ __forceinline__ float bf2f(bf16_t b) {
  union { unsigned short s; bf16_t b; } c; c.b = b;
  union { unsigned u; float f; } a; a.u = ((unsigned)c.s) << 16;
  return a.f;
}

union FragAB { v16bf v; v8bf h[2]; };

// ---------------------------------------------------------------- cast
__global__ void cast_f32_bf16(const float* __restrict__ src,
                              bf16_t* __restrict__ dst, int n) {
  int i = blockIdx.x * blockDim.x + threadIdx.x;
  if (i < n) dst[i] = f2bf(src[i]);
}

// ---------------------------------------------------------------- rope
// X layout [B,S,H,128]; pair index -> (…, d) & (…, d+64)
__global__ void rope_bf16(bf16_t* __restrict__ X, int nheads, int npairs) {
  int idx = blockIdx.x * blockDim.x + threadIdx.x;
  if (idx >= npairs) return;
  int d = idx & 63;
  int s = (idx / (64 * nheads)) % SEQ_;
  size_t base = (size_t)(idx >> 6) * 128 + d;
  float inv = powf(10000.0f, -((float)(2 * d)) / 128.0f);
  float fr  = (float)s * inv;
  float cs  = cosf(fr), sn = sinf(fr);
  float x1 = bf2f(X[base]);
  float x2 = bf2f(X[base + 64]);
  X[base]      = f2bf(x1 * cs - x2 * sn);
  X[base + 64] = f2bf(x2 * cs + x1 * sn);
}

// ---------------------------------------------------------------- GEMM
// Out[M,N] = A[M,K] @ W[N,K]^T + bias ; A,W bf16, acc f32.
// block = 128 thr (4 waves), block tile 64x64, wave tile 16(M)x64(N).
template <bool OUT_F32>
__global__ __launch_bounds__(128) void gemm_wmma(
    const bf16_t* __restrict__ A, const bf16_t* __restrict__ W,
    const float* __restrict__ bias, void* __restrict__ Out,
    int M, int N, int K) {
  const int lane = threadIdx.x & 31;
  const int w    = threadIdx.x >> 5;
  const int hi   = (lane >= 16);
  const int l15  = lane & 15;
  const int n0g  = blockIdx.x * 64;
  const int m0   = blockIdx.y * 64 + w * 16;

  const v8f vzero = {0.f, 0.f, 0.f, 0.f, 0.f, 0.f, 0.f, 0.f};
  v8f acc[4];
#pragma unroll
  for (int nt = 0; nt < 4; ++nt) acc[nt] = vzero;

  const bf16_t* Arow = A + (size_t)(m0 + l15) * K + (hi ? 8 : 0);

  for (int k0 = 0; k0 < K; k0 += 32) {
    __builtin_prefetch((const void*)(Arow + k0 + 256), 0, 1);
    FragAB a;
    a.h[0] = *(const v8bf*)(Arow + k0);
    a.h[1] = *(const v8bf*)(Arow + k0 + 16);
#pragma unroll
    for (int nt = 0; nt < 4; ++nt) {
      const bf16_t* Wrow =
          W + (size_t)(n0g + nt * 16 + l15) * K + k0 + (hi ? 8 : 0);
      FragAB bf;
      bf.h[0] = *(const v8bf*)Wrow;
      bf.h[1] = *(const v8bf*)(Wrow + 16);
      v16bf bb = bf.v;
      acc[nt] = __builtin_amdgcn_wmma_f32_16x16x32_bf16(
          false, a.v, false, bb, (short)0, acc[nt], false, false);
    }
  }
#pragma unroll
  for (int nt = 0; nt < 4; ++nt) {
    int n = n0g + nt * 16 + l15;
    float bv = bias[n];
#pragma unroll
    for (int r = 0; r < 8; ++r) {
      int m = m0 + r + (hi ? 8 : 0);
      float val = acc[nt][r] + bv;
      if (OUT_F32) ((float*)Out)[(size_t)m * N + n] = val;
      else         ((bf16_t*)Out)[(size_t)m * N + n] = f2bf(val);
    }
  }
}

// ---------------------------------------------------------------- flash attn
// grid (S/64, NQH, B), block 128 (4 waves). Q[B,S,NQH,128] K/V[B,S,NKVH,128]
__global__ __launch_bounds__(128) void flash_attn_wmma(
    const bf16_t* __restrict__ Q, const bf16_t* __restrict__ Km,
    const bf16_t* __restrict__ Vm, bf16_t* __restrict__ Ao) {
  const int qs  = blockIdx.x * 64;
  const int h   = blockIdx.y;
  const int b   = blockIdx.z;
  const int kvh = h / (NQH / NKVH);
  const int t    = threadIdx.x;
  const int lane = t & 31;
  const int w    = t >> 5;
  const int hi   = (lane >= 16);
  const int l15  = lane & 15;
  const int m0   = w * 16;
  const float scale = 0.08838834764831845f; // 1/sqrt(128)

  __shared__ float  Sf[64][68];
  __shared__ bf16_t P[64][64];
  __shared__ bf16_t VtT[128][72];   // transposed V tile: [d][key]
  __shared__ float  s_alpha[64];
  __shared__ float  s_l[64];

  const v8f vzero = {0.f, 0.f, 0.f, 0.f, 0.f, 0.f, 0.f, 0.f};
  v8f acc[8];
#pragma unroll
  for (int n2 = 0; n2 < 8; ++n2) acc[n2] = vzero;

  float m_run = -1e30f, l_run = 0.0f;

  const bf16_t* Qbase = Q + (size_t)(b * SEQ_ + qs + m0 + l15) * (NQH * HEAD_) +
                        h * HEAD_ + (hi ? 8 : 0);

  const int kt0 = (qs >= WIN_) ? (qs - WIN_) : 0;
  for (int kt = kt0; kt <= qs; kt += 64) {
    // ---- scores S = Q @ K^T (scaled, masked) -> LDS
#pragma unroll
    for (int nt = 0; nt < 4; ++nt) {
      v8f sacc = vzero;
      const bf16_t* Kbase =
          Km + (size_t)(b * SEQ_ + kt + nt * 16 + l15) * (NKVH * HEAD_) +
          kvh * HEAD_ + (hi ? 8 : 0);
#pragma unroll
      for (int kk = 0; kk < 4; ++kk) {
        FragAB a;
        a.h[0] = *(const v8bf*)(Qbase + kk * 32);
        a.h[1] = *(const v8bf*)(Qbase + kk * 32 + 16);
        FragAB bf;
        bf.h[0] = *(const v8bf*)(Kbase + kk * 32);
        bf.h[1] = *(const v8bf*)(Kbase + kk * 32 + 16);
        v16bf bb = bf.v;
        sacc = __builtin_amdgcn_wmma_f32_16x16x32_bf16(
            false, a.v, false, bb, (short)0, sacc, false, false);
      }
#pragma unroll
      for (int r = 0; r < 8; ++r) {
        int lr = m0 + r + (hi ? 8 : 0);
        int i  = qs + lr;
        int j  = kt + nt * 16 + l15;
        float vv = sacc[r] * scale;
        if (j > i || j < i - WIN_) vv = -1e30f;
        Sf[lr][nt * 16 + l15] = vv;
      }
    }
    __syncthreads();

    if (t < 64) {
      // ---- online softmax bookkeeping, row t
      float mx = m_run;
      for (int j = 0; j < 64; ++j) mx = fmaxf(mx, Sf[t][j]);
      float al = __expf(m_run - mx);
      float sum = 0.f;
      for (int j = 0; j < 64; ++j) {
        float p = __expf(Sf[t][j] - mx);
        P[t][j] = f2bf(p);
        sum += p;
      }
      l_run = l_run * al + sum;
      m_run = mx;
      s_alpha[t] = al;
    } else {
      // ---- stage V tile transposed into LDS
      int t2 = t - 64;
      const bf16_t* Vrow =
          Vm + (size_t)(b * SEQ_ + kt + t2) * (NKVH * HEAD_) + kvh * HEAD_;
#pragma unroll
      for (int c = 0; c < 16; ++c) {
        union { v8bf v; bf16_t e[8]; } u;
        u.v = *(const v8bf*)(Vrow + c * 8);
#pragma unroll
        for (int e = 0; e < 8; ++e) VtT[c * 8 + e][t2] = u.e[e];
      }
    }
    __syncthreads();

    // ---- rescale running O, then O += P @ V
    float alr[8];
#pragma unroll
    for (int r = 0; r < 8; ++r) alr[r] = s_alpha[m0 + r + (hi ? 8 : 0)];
#pragma unroll
    for (int n2 = 0; n2 < 8; ++n2)
#pragma unroll
      for (int r = 0; r < 8; ++r) acc[n2][r] *= alr[r];

#pragma unroll
    for (int kk2 = 0; kk2 < 2; ++kk2) {
      FragAB a;
      a.h[0] = *(const v8bf*)&P[m0 + l15][kk2 * 32 + (hi ? 8 : 0)];
      a.h[1] = *(const v8bf*)&P[m0 + l15][kk2 * 32 + (hi ? 8 : 0) + 16];
#pragma unroll
      for (int n2 = 0; n2 < 8; ++n2) {
        FragAB bb;
        bb.h[0] = *(const v8bf*)&VtT[n2 * 16 + l15][kk2 * 32 + (hi ? 8 : 0)];
        bb.h[1] = *(const v8bf*)&VtT[n2 * 16 + l15][kk2 * 32 + (hi ? 8 : 0) + 16];
        acc[n2] = __builtin_amdgcn_wmma_f32_16x16x32_bf16(
            false, a.v, false, bb.v, (short)0, acc[n2], false, false);
      }
    }
  }

  if (t < 64) s_l[t] = l_run;
  __syncthreads();

#pragma unroll
  for (int n2 = 0; n2 < 8; ++n2) {
#pragma unroll
    for (int r = 0; r < 8; ++r) {
      int lr = m0 + r + (hi ? 8 : 0);
      float val = acc[n2][r] / s_l[lr];
      Ao[(size_t)(b * SEQ_ + qs + lr) * (NQH * HEAD_) + h * HEAD_ + n2 * 16 +
         l15] = f2bf(val);
    }
  }
}

// ---------------------------------------------------------------- launch
extern "C" void kernel_launch(void* const* d_in, const int* in_sizes, int n_in,
                              void* d_out, int out_size, void* d_ws,
                              size_t ws_size, hipStream_t stream) {
  (void)in_sizes; (void)n_in; (void)out_size; (void)ws_size;
  const float* x  = (const float*)d_in[0];
  const float* Wq = (const float*)d_in[1];
  const float* bq = (const float*)d_in[2];
  const float* Wk = (const float*)d_in[3];
  const float* bk = (const float*)d_in[4];
  const float* Wv = (const float*)d_in[5];
  const float* bv = (const float*)d_in[6];
  const float* Wo = (const float*)d_in[7];
  const float* bo = (const float*)d_in[8];

  char* ws = (char*)d_ws;
  size_t off = 0;
  auto carve = [&](size_t nelem) {
    bf16_t* p = (bf16_t*)(ws + off);
    off += nelem * sizeof(bf16_t);
    return p;
  };
  bf16_t* xb  = carve((size_t)MTOT_ * EMB_);          // 16 MB
  bf16_t* Wqb = carve((size_t)EMB_ * EMB_);           //  8 MB
  bf16_t* Wkb = carve((size_t)NKVH * HEAD_ * EMB_);   //  2 MB
  bf16_t* Wvb = carve((size_t)NKVH * HEAD_ * EMB_);   //  2 MB
  bf16_t* Wob = carve((size_t)EMB_ * EMB_);           //  8 MB
  bf16_t* Qb  = carve((size_t)MTOT_ * NQH * HEAD_);   // 16 MB
  bf16_t* Kb  = carve((size_t)MTOT_ * NKVH * HEAD_);  //  4 MB
  bf16_t* Vb  = carve((size_t)MTOT_ * NKVH * HEAD_);  //  4 MB
  bf16_t* Ab  = carve((size_t)MTOT_ * NQH * HEAD_);   // 16 MB

  auto castN = [&](const float* s, bf16_t* d, int n) {
    cast_f32_bf16<<<(n + 255) / 256, 256, 0, stream>>>(s, d, n);
  };
  castN(x,  xb,  MTOT_ * EMB_);
  castN(Wq, Wqb, EMB_ * EMB_);
  castN(Wk, Wkb, NKVH * HEAD_ * EMB_);
  castN(Wv, Wvb, NKVH * HEAD_ * EMB_);
  castN(Wo, Wob, EMB_ * EMB_);

  // Q/K/V projections (bf16 out)
  gemm_wmma<false><<<dim3(EMB_ / 64, MTOT_ / 64), 128, 0, stream>>>(
      xb, Wqb, bq, Qb, MTOT_, EMB_, EMB_);
  gemm_wmma<false><<<dim3((NKVH * HEAD_) / 64, MTOT_ / 64), 128, 0, stream>>>(
      xb, Wkb, bk, Kb, MTOT_, NKVH * HEAD_, EMB_);
  gemm_wmma<false><<<dim3((NKVH * HEAD_) / 64, MTOT_ / 64), 128, 0, stream>>>(
      xb, Wvb, bv, Vb, MTOT_, NKVH * HEAD_, EMB_);

  // RoPE in place
  {
    int np_q = BATCH_ * SEQ_ * NQH * 64;
    int np_k = BATCH_ * SEQ_ * NKVH * 64;
    rope_bf16<<<(np_q + 255) / 256, 256, 0, stream>>>(Qb, NQH, np_q);
    rope_bf16<<<(np_k + 255) / 256, 256, 0, stream>>>(Kb, NKVH, np_k);
  }

  // fused sliding-window GQA attention
  flash_attn_wmma<<<dim3(SEQ_ / 64, NQH, BATCH_), 128, 0, stream>>>(Qb, Kb, Vb,
                                                                    Ab);

  // output projection (f32 out)
  gemm_wmma<true><<<dim3(EMB_ / 64, MTOT_ / 64), 128, 0, stream>>>(
      Ab, Wob, bo, (float*)d_out, MTOT_, EMB_, EMB_);
}
